// CWRRTWindowCell_73358041415701
// MI455X (gfx1250) — matrix operand order repaired and checked by reading, not compiled, not run on hardware
//
#include <hip/hip_runtime.h>
#include <hip/hip_bf16.h>
#include <math.h>

// ---------------- problem dimensions ----------------
constexpr int Bb = 4, Tt = 1024, Oo = 128, Cc = 1024, Hh = 16, Ll = 4;
constexpr int Vv = 32000, Ss = 4096, Dd = 64;
constexpr int OT = Oo + Tt; // 1152 (mem + tokens)

typedef __attribute__((ext_vector_type(16))) __bf16 v16bf;
typedef __attribute__((ext_vector_type(8)))  float  v8f;
typedef __attribute__((ext_vector_type(4)))  unsigned int u32x4;
typedef __attribute__((ext_vector_type(8)))  int i32x8;
typedef __attribute__((ext_vector_type(4)))  int i32x4;

#if defined(__has_builtin)
#if __has_builtin(__builtin_amdgcn_tensor_load_to_lds)
#define HAVE_TDM 1
#endif
#endif

__device__ inline v8f wmma_bf16(v16bf a, v16bf b, v8f c) {
  // D(16x16,f32) = A(16x32,bf16) * B(32x16,bf16) + C
  return __builtin_amdgcn_wmma_f32_16x16x32_bf16(false, a, false, b, (short)0, c,
                                                 false, false);
}
__device__ inline v8f zero8() {
  v8f z = {0.f, 0.f, 0.f, 0.f, 0.f, 0.f, 0.f, 0.f};
  return z;
}

__device__ inline uint2 pack4bf(float4 v) {
  union { __bf16 h[4]; uint2 u; } x;
  x.h[0] = (__bf16)v.x; x.h[1] = (__bf16)v.y;
  x.h[2] = (__bf16)v.z; x.h[3] = (__bf16)v.w;
  return x.u;
}
__device__ inline uint4 pack8bf(float4 a, float4 b) {
  union { __bf16 h[8]; uint4 u; } x;
  x.h[0] = (__bf16)a.x; x.h[1] = (__bf16)a.y;
  x.h[2] = (__bf16)a.z; x.h[3] = (__bf16)a.w;
  x.h[4] = (__bf16)b.x; x.h[5] = (__bf16)b.y;
  x.h[6] = (__bf16)b.z; x.h[7] = (__bf16)b.w;
  return x.u;
}

// A fragment (16x32 bf16) from row-major LDS tile, leading dim ld (mult of 8).
// ISA 7.12.2: m=lane&15, half=lane>>4; e<8 -> K=e+8*half ; e>=8 -> K=e+8+8*half
// => per lane two contiguous 8-element runs: 2x ds_load_b128.
__device__ inline v16bf load_A(const __bf16* base, int ld, int lane) {
  int m = lane & 15, hf = lane >> 4;
  union { v16bf v; uint4 q[2]; } u;
  u.q[0] = *(const uint4*)&base[m * ld + hf * 8];
  u.q[1] = *(const uint4*)&base[m * ld + 16 + hf * 8];
  return u.v;
}
// B fragment (32x16 bf16) from a COLUMN-major (n-major) LDS tile:
// tile[n][k], leading dim ld (mult of 8). lane n=lane&15, e<8 -> K=e+8*half ; e>=8 -> K=e+8+8*half
// => per lane two contiguous 8-element runs: 2x ds_load_b128.
__device__ inline v16bf load_Bt(const __bf16* base, int ld, int lane) {
  int n = lane & 15, hf = lane >> 4;
  union { v16bf v; uint4 q[2]; } u;
  u.q[0] = *(const uint4*)&base[n * ld + hf * 8];
  u.q[1] = *(const uint4*)&base[n * ld + 16 + hf * 8];
  return u.v;
}

__device__ inline float gelu_f(float x) {
  const float k0 = 0.7978845608028654f, k1 = 0.044715f;
  return 0.5f * x * (1.f + tanhf(k0 * (x + k1 * x * x * x)));
}
__device__ inline float sigmoid_f(float x) { return 1.f / (1.f + __expf(-x)); }

#ifdef HAVE_TDM
// Issue a Tensor-Data-Mover load of a 32x64 f32 tile (row stride N) into LDS.
__device__ inline void tdm_issue_tile(const float* gp, unsigned lds_off, int N,
                                      int K) {
  unsigned long long ga = (unsigned long long)(const void*)gp;
  u32x4 g0;
  g0[0] = 1u;                                    // count=1 (valid), user mode
  g0[1] = lds_off;                               // lds_addr (bytes)
  g0[2] = (unsigned int)ga;                      // global_addr[31:0]
  g0[3] = (unsigned int)((ga >> 32) & 0x1FFFFFFull) | (2u << 30);  // + type=2
  i32x8 g1;
  g1[0] = (2 << 16);                             // data_size=4B, no mcast mask
  g1[1] = (int)((unsigned)(N & 0xFFFF) << 16);   // tensor_dim0[15:0]
  g1[2] = (int)(((unsigned)N >> 16) | ((unsigned)(K & 0xFFFF) << 16));
  g1[3] = (int)((((unsigned)K >> 16) & 0xFFFFu) | (64u << 16));  // tile_dim0=64
  g1[4] = 32;                                    // tile_dim1=32, tile_dim2=0
  g1[5] = N;                                     // tensor_dim0_stride[31:0]
  g1[6] = 0;
  g1[7] = 0;
  i32x4 gz = {0, 0, 0, 0};
#if defined(__clang_major__) && (__clang_major__ >= 23)
  i32x8 gz8 = {0, 0, 0, 0, 0, 0, 0, 0};
  __builtin_amdgcn_tensor_load_to_lds(g0, g1, gz, gz, gz8, 0);
#else
  __builtin_amdgcn_tensor_load_to_lds(g0, g1, gz, gz, 0);
#endif
}
#endif

// ---------------- small elementwise kernels ----------------

__global__ void sumterm_kernel(const float* __restrict__ ssum,
                               const float* __restrict__ sum_w,
                               const float* __restrict__ alpha_p,
                               float* __restrict__ st) {
  int idx = blockIdx.x * blockDim.x + threadIdx.x;
  if (idx >= Bb * Cc) return;
  int c = idx % Cc, b = idx / Cc;
  float acc = 0.f;
  for (int k = 0; k < Cc; ++k) acc += ssum[b * Cc + k] * sum_w[k * Cc + c];
  st[idx] = acc * sigmoid_f(alpha_p[c]);
}

__global__ void embed_kernel(const int* __restrict__ tokens,
                             const int* __restrict__ pos_off,
                             const float* __restrict__ tok_emb,
                             const float* __restrict__ pos_emb,
                             const float* __restrict__ st,
                             float* __restrict__ x) {
  long idx = (long)blockIdx.x * blockDim.x + threadIdx.x;
  if (idx >= (long)Bb * Tt * Cc) return;
  int c = idx % Cc;
  int t = (idx / Cc) % Tt;
  int b = idx / ((long)Cc * Tt);
  int tok = tokens[b * Tt + t];
  int p = t + pos_off[0];
  p = p < 0 ? 0 : (p > Ss - 1 ? Ss - 1 : p);
  x[idx] = tok_emb[(long)tok * Cc + c] + pos_emb[(long)p * Cc + c] + st[b * Cc + c];
}

__global__ void __launch_bounds__(256) ln_kernel(const float* __restrict__ in,
                                                 const float* __restrict__ sc,
                                                 const float* __restrict__ bi,
                                                 float* __restrict__ out) {
  __shared__ float red[256];
  long row = blockIdx.x;
  const float* xr = in + row * Cc;
  int tid = threadIdx.x;
  float s = 0.f;
  for (int c = tid; c < Cc; c += 256) s += xr[c];
  red[tid] = s;
  __syncthreads();
  for (int st = 128; st > 0; st >>= 1) {
    if (tid < st) red[tid] += red[tid + st];
    __syncthreads();
  }
  float mu = red[0] / Cc;
  __syncthreads();
  float v = 0.f;
  for (int c = tid; c < Cc; c += 256) {
    float d = xr[c] - mu;
    v += d * d;
  }
  red[tid] = v;
  __syncthreads();
  for (int st = 128; st > 0; st >>= 1) {
    if (tid < st) red[tid] += red[tid + st];
    __syncthreads();
  }
  float rstd = rsqrtf(red[0] / Cc + 1e-6f);
  for (int c = tid; c < Cc; c += 256)
    out[row * Cc + c] = (xr[c] - mu) * rstd * sc[c] + bi[c];
}

__global__ void kv_assemble_kernel(const float* __restrict__ m2,
                                   const float* __restrict__ x,
                                   float* __restrict__ kv) {
  long idx = (long)blockIdx.x * blockDim.x + threadIdx.x;
  if (idx >= (long)Bb * OT * Cc) return;
  int c = idx % Cc;
  int s = (idx / Cc) % OT;
  int b = idx / ((long)Cc * OT);
  kv[idx] = (s < Oo) ? m2[((long)b * Oo + s) * Cc + c]
                     : x[((long)b * Tt + (s - Oo)) * Cc + c];
}

__global__ void mask_kernel(const float* __restrict__ x,
                            const int* __restrict__ tokens,
                            float* __restrict__ xm) {
  long idx = (long)blockIdx.x * blockDim.x + threadIdx.x;
  if (idx >= (long)Bb * Tt * Cc) return;
  int t = (idx / Cc) % Tt;
  int b = idx / ((long)Cc * Tt);
  xm[idx] = (tokens[b * Tt + t] != 0) ? x[idx] : 0.f;
}

__global__ void newmem_kernel(const float* __restrict__ xf, float* __restrict__ nm) {
  long idx = (long)blockIdx.x * blockDim.x + threadIdx.x;
  if (idx >= (long)Bb * Oo * Cc) return;
  int c = idx % Cc;
  int o = (idx / Cc) % Oo;
  int b = idx / ((long)Cc * Oo);
  nm[idx] = xf[((long)b * Tt + (Tt - Oo + o)) * Cc + c];
}

__global__ void ssum_kernel(const float* __restrict__ xm,
                            const int* __restrict__ tokens,
                            const float* __restrict__ ssum,
                            const float* __restrict__ lam_p,
                            float* __restrict__ outs) {
  int idx = blockIdx.x * blockDim.x + threadIdx.x;
  if (idx >= Bb * Cc) return;
  int c = idx % Cc, b = idx / Cc;
  float s = 0.f;
  for (int t = 0; t < Tt; ++t) s += xm[((long)b * Tt + t) * Cc + c];
  float cnt = 0.f;
  for (int t = 0; t < Tt; ++t) cnt += (tokens[b * Tt + t] != 0) ? 1.f : 0.f;
  float summary = s / (cnt + 1e-6f);
  float lam = sigmoid_f(lam_p[c]);
  outs[idx] = ssum[idx] * lam + summary * (1.f - lam);
}

// ---------------- WMMA GEMM: out = act(A@W + bias) (+resid) ----------------
// A: MxK f32, W: KxN f32, M % 64 == 0 for all call sites.
// W tiles are DMA'd f32 into a double-buffered LDS region by the Tensor Data
// Mover (issued by wave 0, overlapped with compute), then transposed+converted
// to an n-major bf16 tile so B fragments are 2x ds_load_b128.
__global__ void __launch_bounds__(128) gemm_wmma_kernel(
    const float* __restrict__ A, const float* __restrict__ W,
    const float* __restrict__ bias, const float* __restrict__ resid,
    float* __restrict__ out, int M, int N, int K, int act) {
  constexpr int LDA = 40, LDT = 40;
  __shared__ __align__(16) __bf16 As[64 * LDA];    // row-major 64x32
  __shared__ __align__(16) __bf16 BsT[64 * LDT];   // n-major 64x32
#ifdef HAVE_TDM
  __shared__ __align__(16) float Bf32[2][32 * 64];
#endif
  int tid = threadIdx.x;
  int lane = tid & 31, wave = tid >> 5;
  int row0 = blockIdx.y * 64;
  int col0 = blockIdx.x * 64;
  int wr = (wave >> 1) * 32;
  int wc = (wave & 1) * 32;
  v8f c00 = zero8(), c01 = zero8(), c10 = zero8(), c11 = zero8();

#ifdef HAVE_TDM
  unsigned bf_off0 = (unsigned)(unsigned long long)(void*)&Bf32[0][0];
  unsigned bf_off1 = (unsigned)(unsigned long long)(void*)&Bf32[1][0];
  if (wave == 0) tdm_issue_tile(W + col0, bf_off0, N, K);  // k0 = 0
#endif

  for (int k0 = 0; k0 < K; k0 += 32) {
    // --- stage A tile: 64x32 f32 -> bf16 (float4 loads, packed b64 stores) ---
#pragma unroll
    for (int i = 0; i < 4; ++i) {
      int idx = i * 128 + tid;     // 512 float4 groups
      int r = idx >> 3;            // row 0..63
      int c4 = (idx & 7) * 4;      // k 0,4,..,28
      int gr = row0 + r;
      if (gr > M - 1) gr = M - 1;  // branch-free (M % 64 == 0 anyway)
      float4 va = *(const float4*)(A + (long)gr * K + k0 + c4);
      *(uint2*)&As[r * LDA + c4] = pack4bf(va);
    }
    if (k0 + 32 < K)  // prefetch next A tile (global_prefetch_b8)
      __builtin_prefetch(A + (long)(row0 + (tid >> 1)) * K + k0 + 32, 0, 1);

#ifdef HAVE_TDM
    int buf = (k0 >> 5) & 1;
    if (wave == 0) {
      if (k0 + 32 < K) {  // issue next DMA, then wait for current (in-order)
        tdm_issue_tile(W + (long)(k0 + 32) * N + col0, buf ? bf_off0 : bf_off1,
                       N, K);
        __builtin_amdgcn_s_wait_tensorcnt(1);
      } else {
        __builtin_amdgcn_s_wait_tensorcnt(0);
      }
    }
    __syncthreads();
    // transpose+convert DMA'd f32 tile -> n-major bf16 BsT (4x4 block/thread)
    {
      const float* Bt = &Bf32[buf][0];
      int n4 = (tid & 15) * 4;     // n 0,4,..,60
      int k4 = (tid >> 4) * 4;     // k 0,4,..,28
      float4 r0 = *(const float4*)&Bt[(k4 + 0) * 64 + n4];
      float4 r1 = *(const float4*)&Bt[(k4 + 1) * 64 + n4];
      float4 r2 = *(const float4*)&Bt[(k4 + 2) * 64 + n4];
      float4 r3 = *(const float4*)&Bt[(k4 + 3) * 64 + n4];
      *(uint2*)&BsT[(n4 + 0) * LDT + k4] = pack4bf(float4{r0.x, r1.x, r2.x, r3.x});
      *(uint2*)&BsT[(n4 + 1) * LDT + k4] = pack4bf(float4{r0.y, r1.y, r2.y, r3.y});
      *(uint2*)&BsT[(n4 + 2) * LDT + k4] = pack4bf(float4{r0.z, r1.z, r2.z, r3.z});
      *(uint2*)&BsT[(n4 + 3) * LDT + k4] = pack4bf(float4{r0.w, r1.w, r2.w, r3.w});
    }
#else
    // fallback: stage W tile directly with a 4x4 register transpose
    {
      int n4 = (tid & 15) * 4;
      int k4 = (tid >> 4) * 4;
      const float* Wt = W + (long)(k0 + k4) * N + col0 + n4;
      float4 r0 = *(const float4*)(Wt + 0 * N);
      float4 r1 = *(const float4*)(Wt + 1 * N);
      float4 r2 = *(const float4*)(Wt + 2 * N);
      float4 r3 = *(const float4*)(Wt + 3 * N);
      *(uint2*)&BsT[(n4 + 0) * LDT + k4] = pack4bf(float4{r0.x, r1.x, r2.x, r3.x});
      *(uint2*)&BsT[(n4 + 1) * LDT + k4] = pack4bf(float4{r0.y, r1.y, r2.y, r3.y});
      *(uint2*)&BsT[(n4 + 2) * LDT + k4] = pack4bf(float4{r0.z, r1.z, r2.z, r3.z});
      *(uint2*)&BsT[(n4 + 3) * LDT + k4] = pack4bf(float4{r0.w, r1.w, r2.w, r3.w});
    }
#endif
    __syncthreads();
    v16bf a0 = load_A(&As[wr * LDA], LDA, lane);
    v16bf a1 = load_A(&As[(wr + 16) * LDA], LDA, lane);
    v16bf b0 = load_Bt(&BsT[wc * LDT], LDT, lane);
    v16bf b1 = load_Bt(&BsT[(wc + 16) * LDT], LDT, lane);
    c00 = wmma_bf16(a0, b0, c00);
    c01 = wmma_bf16(a0, b1, c01);
    c10 = wmma_bf16(a1, b0, c10);
    c11 = wmma_bf16(a1, b1, c11);
    __syncthreads();
  }

  int n = lane & 15, hf = lane >> 4;
  auto st = [&](v8f cf, int ro, int co) {
#pragma unroll
    for (int r = 0; r < 8; ++r) {
      int gr = row0 + wr + ro + r + hf * 8;
      int gc = col0 + wc + co + n;
      float v = cf[r];
      if (bias) v += bias[gc];
      if (act == 1) v = gelu_f(v);
      if (resid) v += resid[(long)gr * N + gc];
      out[(long)gr * N + gc] = v;
    }
  };
  st(c00, 0, 0);
  st(c01, 0, 16);
  st(c10, 16, 0);
  st(c11, 16, 16);
}

// ---------------- flash attention (one wave per 16-query tile) ----------------
__global__ void __launch_bounds__(128) attn_kernel(const float* __restrict__ Q,
                                                   const float* __restrict__ Kb,
                                                   const float* __restrict__ Vb,
                                                   const int* __restrict__ tokens,
                                                   float* __restrict__ outp) {
  // per-wave: q 16x64, k 32x64 (row-major), vT 64x40 (d-major), p 16x32
  constexpr int LDV = 40;
  constexpr int WSM = 1024 + 2048 + 64 * LDV + 512;
  __shared__ __align__(16) __bf16 smem[4 * WSM];
  int lane = threadIdx.x & 31;
  int wave = threadIdx.x >> 5;
  __bf16* q_s = smem + wave * WSM;
  __bf16* k_s = q_s + 1024;
  __bf16* vT_s = k_s + 2048;
  __bf16* p_s = vT_s + 64 * LDV;
  int hh = blockIdx.y, b = blockIdx.z;
  int q0 = (blockIdx.x * 4 + wave) * 16;
  const float scale = 0.125f; // 1/sqrt(64)

  { // stage Q tile (16x64) -> bf16 LDS, vectorized
    int r = lane >> 1, cb = (lane & 1) * 32;
    const float* qp = Q + ((long)(b * Tt + q0 + r)) * Cc + hh * Dd + cb;
#pragma unroll
    for (int d = 0; d < 32; d += 8)
      *(uint4*)&q_s[r * 64 + cb + d] =
          pack8bf(*(const float4*)(qp + d), *(const float4*)(qp + d + 4));
  }
  asm volatile("s_wait_dscnt 0" ::: "memory");
  v16bf qa0 = load_A(q_s, 64, lane);       // d 0..31
  v16bf qa1 = load_A(q_s + 32, 64, lane);  // d 32..63

  v8f o0 = zero8(), o1 = zero8(), o2 = zero8(), o3 = zero8();
  float mrow[8], lrow[8];
#pragma unroll
  for (int r = 0; r < 8; ++r) {
    mrow[r] = -3e38f;
    lrow[r] = 0.f;
  }
  int n = lane & 15, hf = lane >> 4;
  int kend = Oo + q0 + 16; // causal upper bound for this tile

  for (int kt0 = 0; kt0 < kend; kt0 += 32) {
    { // stage K tile (32x64 row-major): lane = key row; clamp OOB rows
      // (rows >= OT are always causally masked, data value irrelevant)
      int kg = kt0 + lane;
      if (kg > OT - 1) kg = OT - 1;
      const float* kp = Kb + ((long)(b * OT + kg)) * Cc + hh * Dd;
#pragma unroll
      for (int d = 0; d < 64; d += 8)
        *(uint4*)&k_s[lane * 64 + d] =
            pack8bf(*(const float4*)(kp + d), *(const float4*)(kp + d + 4));
    }
    { // stage V tile transposed (vT[d][key]); per-lane 4x4 register transposes
#pragma unroll
      for (int i = 0; i < 4; ++i) {
        int bid = i * 32 + lane;       // 128 4x4 blocks
        int kq4 = (bid & 7) * 4;       // key 0,4,..,28
        int d4 = (bid >> 3) * 4;       // d 0,4,..,60
        int g0i = kt0 + kq4;
        int g0c = g0i + 0 > OT - 1 ? OT - 1 : g0i + 0;
        int g1c = g0i + 1 > OT - 1 ? OT - 1 : g0i + 1;
        int g2c = g0i + 2 > OT - 1 ? OT - 1 : g0i + 2;
        int g3c = g0i + 3 > OT - 1 ? OT - 1 : g0i + 3;
        long hb = (long)b * OT;
        float4 r0 = *(const float4*)(Vb + (hb + g0c) * Cc + hh * Dd + d4);
        float4 r1 = *(const float4*)(Vb + (hb + g1c) * Cc + hh * Dd + d4);
        float4 r2 = *(const float4*)(Vb + (hb + g2c) * Cc + hh * Dd + d4);
        float4 r3 = *(const float4*)(Vb + (hb + g3c) * Cc + hh * Dd + d4);
        *(uint2*)&vT_s[(d4 + 0) * LDV + kq4] = pack4bf(float4{r0.x, r1.x, r2.x, r3.x});
        *(uint2*)&vT_s[(d4 + 1) * LDV + kq4] = pack4bf(float4{r0.y, r1.y, r2.y, r3.y});
        *(uint2*)&vT_s[(d4 + 2) * LDV + kq4] = pack4bf(float4{r0.z, r1.z, r2.z, r3.z});
        *(uint2*)&vT_s[(d4 + 3) * LDV + kq4] = pack4bf(float4{r0.w, r1.w, r2.w, r3.w});
      }
    }
    asm volatile("s_wait_dscnt 0" ::: "memory");

    // scores S(16x32) = Q(16x64) @ K^T(64x32); k_s row-major => contiguous d
    v8f c0 = zero8(), c1 = zero8();
    c0 = wmma_bf16(qa0, load_Bt(&k_s[0 * 64 + 0], 64, lane), c0);
    c0 = wmma_bf16(qa1, load_Bt(&k_s[0 * 64 + 32], 64, lane), c0);
    c1 = wmma_bf16(qa0, load_Bt(&k_s[16 * 64 + 0], 64, lane), c1);
    c1 = wmma_bf16(qa1, load_Bt(&k_s[16 * 64 + 32], 64, lane), c1);

    int kg0 = kt0 + n, kg1 = kt0 + 16 + n;
    int tr0 = kg0 - Oo, tr1 = kg1 - Oo;
    bool tok0 = (tr0 < 0) || (tr0 < Tt && tokens[b * Tt + tr0] != 0);
    bool tok1 = (tr1 < 0) || (tr1 < Tt && tokens[b * Tt + tr1] != 0);

    float s0a[8], s1a[8];
#pragma unroll
    for (int r = 0; r < 8; ++r) {
      int qg = q0 + r + hf * 8;
      bool a0 = (tr0 < 0) || (tr0 <= qg && tok0);
      bool a1 = (tr1 < 0) || (tr1 <= qg && tok1);
      s0a[r] = a0 ? c0[r] * scale : -1e9f;
      s1a[r] = a1 ? c1[r] * scale : -1e9f;
    }
#pragma unroll
    for (int r = 0; r < 8; ++r) {
      float rm = fmaxf(s0a[r], s1a[r]);
      rm = fmaxf(rm, __shfl_xor(rm, 8, 16));
      rm = fmaxf(rm, __shfl_xor(rm, 4, 16));
      rm = fmaxf(rm, __shfl_xor(rm, 2, 16));
      rm = fmaxf(rm, __shfl_xor(rm, 1, 16));
      float mn = fmaxf(mrow[r], rm);
      float al = __expf(mrow[r] - mn);
      mrow[r] = mn;
      float p0 = (s0a[r] < -1e8f) ? 0.f : __expf(s0a[r] - mn);
      float p1 = (s1a[r] < -1e8f) ? 0.f : __expf(s1a[r] - mn);
      float rs = p0 + p1;
      rs += __shfl_xor(rs, 8, 16);
      rs += __shfl_xor(rs, 4, 16);
      rs += __shfl_xor(rs, 2, 16);
      rs += __shfl_xor(rs, 1, 16);
      lrow[r] = lrow[r] * al + rs;
      o0[r] *= al;
      o1[r] *= al;
      o2[r] *= al;
      o3[r] *= al;
      int m = r + hf * 8;
      p_s[m * 32 + n] = (__bf16)p0;
      p_s[m * 32 + 16 + n] = (__bf16)p1;
    }
    asm volatile("s_wait_dscnt 0" ::: "memory");

    // O(16x64) += P(16x32) @ V(32x64); vT_s d-major => contiguous keys
    v16bf pa = load_A(p_s, 32, lane);
    o0 = wmma_bf16(pa, load_Bt(&vT_s[0 * LDV], LDV, lane), o0);
    o1 = wmma_bf16(pa, load_Bt(&vT_s[16 * LDV], LDV, lane), o1);
    o2 = wmma_bf16(pa, load_Bt(&vT_s[32 * LDV], LDV, lane), o2);
    o3 = wmma_bf16(pa, load_Bt(&vT_s[48 * LDV], LDV, lane), o3);
  }

#pragma unroll
  for (int r = 0; r < 8; ++r) {
    float inv = 1.f / lrow[r];
    int tg = q0 + r + hf * 8;
    float* op = outp + ((long)(b * Tt + tg)) * Cc + hh * Dd;
    op[n] = o0[r] * inv;
    op[16 + n] = o1[r] * inv;
    op[32 + n] = o2[r] * inv;
    op[48 + n] = o3[r] * inv;
  }
}

// ---------------- host orchestration ----------------
extern "C" void kernel_launch(void* const* d_in, const int* in_sizes, int n_in,
                              void* d_out, int out_size, void* d_ws, size_t ws_size,
                              hipStream_t stream) {
  (void)in_sizes; (void)n_in; (void)out_size; (void)ws_size;
  const int*   tokens   = (const int*)d_in[0];
  const int*   pos_off  = (const int*)d_in[1];
  const float* mem      = (const float*)d_in[2];
  const float* ssum     = (const float*)d_in[3];
  const float* tok_emb  = (const float*)d_in[4];
  const float* pos_emb  = (const float*)d_in[5];
  const float* mem_w    = (const float*)d_in[6];
  const float* mem_b    = (const float*)d_in[7];
  const float* memln_s  = (const float*)d_in[8];
  const float* memln_b  = (const float*)d_in[9];
  const float* alpha_p  = (const float*)d_in[10];
  const float* sum_w    = (const float*)d_in[11];
  const float* ln1_s    = (const float*)d_in[12];
  const float* ln1_b    = (const float*)d_in[13];
  const float* Wq       = (const float*)d_in[14];
  const float* bq       = (const float*)d_in[15];
  const float* Wk       = (const float*)d_in[16];
  const float* bk       = (const float*)d_in[17];
  const float* Wv       = (const float*)d_in[18];
  const float* bv       = (const float*)d_in[19];
  const float* Wo       = (const float*)d_in[20];
  const float* bo       = (const float*)d_in[21];
  const float* ln2_s    = (const float*)d_in[22];
  const float* ln2_b    = (const float*)d_in[23];
  const float* W1       = (const float*)d_in[24];
  const float* b1       = (const float*)d_in[25];
  const float* W2       = (const float*)d_in[26];
  const float* b2       = (const float*)d_in[27];
  const float* fln_s    = (const float*)d_in[28];
  const float* fln_b    = (const float*)d_in[29];
  const float* head_w   = (const float*)d_in[30];
  const float* head_b   = (const float*)d_in[31];
  const float* lam_p    = (const float*)d_in[32];

  float* out_mem  = (float*)d_out;                          // B*O*C
  float* out_ssum = out_mem + (size_t)Bb * Oo * Cc;         // B*C
  float* logits   = out_ssum + (size_t)Bb * Cc;             // B*T*V

  float* wp = (float*)d_ws;
  auto carve = [&](size_t nelem) { float* r = wp; wp += nelem; return r; };
  float* x   = carve((size_t)Bb * Tt * Cc);
  float* h   = carve((size_t)Bb * Tt * Cc);
  float* q   = carve((size_t)Bb * Tt * Cc);
  float* att = carve((size_t)Bb * Tt * Cc);
  float* xm  = carve((size_t)Bb * Tt * Cc);
  float* kv  = carve((size_t)Bb * OT * Cc);
  float* kb  = carve((size_t)Bb * OT * Cc);
  float* vb  = carve((size_t)Bb * OT * Cc);
  float* ffn = carve((size_t)Bb * Tt * 4 * Cc);
  float* m2  = carve((size_t)Bb * Oo * Cc);
  float* mt  = carve((size_t)Bb * Oo * Cc);
  float* st  = carve((size_t)Bb * Cc);

  const int EL = 256;
  auto gemm = [&](const float* A, const float* W, const float* bias,
                  const float* resid, float* out, int M, int N, int K, int act) {
    dim3 grid(N / 64, (M + 63) / 64);
    gemm_wmma_kernel<<<grid, 128, 0, stream>>>(A, W, bias, resid, out, M, N, K, act);
  };

  sumterm_kernel<<<(Bb * Cc + EL - 1) / EL, EL, 0, stream>>>(ssum, sum_w, alpha_p, st);
  embed_kernel<<<((long)Bb * Tt * Cc + EL - 1) / EL, EL, 0, stream>>>(
      tokens, pos_off, tok_emb, pos_emb, st, x);
  gemm(mem, mem_w, mem_b, nullptr, mt, Bb * Oo, Cc, Cc, 0);
  ln_kernel<<<Bb * Oo, 256, 0, stream>>>(mt, memln_s, memln_b, m2);

  for (int i = 0; i < Ll; ++i) {
    const size_t wo = (size_t)i * Cc * Cc;
    ln_kernel<<<Bb * Tt, 256, 0, stream>>>(x, ln1_s + i * Cc, ln1_b + i * Cc, h);
    gemm(h, Wq + wo, bq + i * Cc, nullptr, q, Bb * Tt, Cc, Cc, 0);
    kv_assemble_kernel<<<((long)Bb * OT * Cc + EL - 1) / EL, EL, 0, stream>>>(m2, x, kv);
    gemm(kv, Wk + wo, bk + i * Cc, nullptr, kb, Bb * OT, Cc, Cc, 0);
    gemm(kv, Wv + wo, bv + i * Cc, nullptr, vb, Bb * OT, Cc, Cc, 0);
    attn_kernel<<<dim3(Tt / 16 / 4, Hh, Bb), 128, 0, stream>>>(q, kb, vb, tokens, att);
    gemm(att, Wo + wo, bo + i * Cc, x, x, Bb * Tt, Cc, Cc, 0);
    ln_kernel<<<Bb * Tt, 256, 0, stream>>>(x, ln2_s + i * Cc, ln2_b + i * Cc, h);
    gemm(h, W1 + (size_t)i * Cc * 4 * Cc, b1 + i * 4 * Cc, nullptr, ffn,
         Bb * Tt, 4 * Cc, Cc, 1);
    gemm(ffn, W2 + (size_t)i * 4 * Cc * Cc, b2 + i * Cc, x, x,
         Bb * Tt, Cc, 4 * Cc, 0);
  }

  ln_kernel<<<Bb * Tt, 256, 0, stream>>>(x, fln_s, fln_b, h);
  newmem_kernel<<<((long)Bb * Oo * Cc + EL - 1) / EL, EL, 0, stream>>>(h, out_mem);
  mask_kernel<<<((long)Bb * Tt * Cc + EL - 1) / EL, EL, 0, stream>>>(h, tokens, xm);
  ssum_kernel<<<(Bb * Cc + EL - 1) / EL, EL, 0, stream>>>(xm, tokens, ssum, lam_p,
                                                          out_ssum);
  gemm(xm, head_w, head_b, nullptr, logits, Bb * Tt, Vv, Cc, 0);
}
